// LSTM_13271448945142
// MI455X (gfx1250) — matrix operand order repaired and checked by reading, not compiled, not run on hardware
//
#include <hip/hip_runtime.h>
#include <hip/hip_bf16.h>
#include <stdint.h>
#include <stddef.h>

#define B_  256
#define T_  128
#define H_  1024
#define C_  10
#define G4H (4 * H_)

#define KCHUNK 256                       // K elements staged per LDS chunk
#define NCHUNK (H_ / KCHUNK)             // 4 chunks
#define AROW   (KCHUNK + 8)              // 264 bf16 row pitch: 528B = 132 dw,
                                         // 132 % 64 = 4 -> staggered banks

typedef __attribute__((ext_vector_type(8)))  __bf16 v8bf;
typedef __attribute__((ext_vector_type(16))) __bf16 v16bf;
typedef __attribute__((ext_vector_type(8)))  float  v8f;
typedef __attribute__((ext_vector_type(4)))  int    v4i;

typedef __attribute__((address_space(1))) v4i as1_v4i;  // global 16B vector
typedef __attribute__((address_space(3))) v4i as3_v4i;  // LDS 16B vector

// --- CDNA5 async global->LDS copy (ASYNCcnt DMA), with safe fallback -------
#if __has_builtin(__builtin_amdgcn_global_load_async_to_lds_b128)
#define ASYNC_COPY_B128(gp, lp)                                              \
    __builtin_amdgcn_global_load_async_to_lds_b128((as1_v4i*)(gp),           \
                                                   (as3_v4i*)(lp), 0, 0)
#else
#define ASYNC_COPY_B128(gp, lp) (*(v8bf*)(lp) = *(const v8bf*)(gp))
#endif

#if __has_builtin(__builtin_amdgcn_s_wait_asynccnt)
#define ASYNC_WAIT() __builtin_amdgcn_s_wait_asynccnt(0)
#else
#define ASYNC_WAIT() asm volatile("s_wait_asynccnt 0x0" ::: "memory")
#endif

__device__ __forceinline__ float sigmoidf_(float x) {
    return 1.0f / (1.0f + __expf(-x));
}

// ---------------------------------------------------------------------------
// One-time packing: fuse 4 gate weight matrices [H,H] fp32 -> Wb [4H,H] bf16.
// Gate order matches reference concat: g (tanh), i, f, o.
// ---------------------------------------------------------------------------
__global__ void pack_wh(const float* __restrict__ Wg, const float* __restrict__ Wi,
                        const float* __restrict__ Wf, const float* __restrict__ Wo,
                        __bf16* __restrict__ Wb) {
    int idx = blockIdx.x * blockDim.x + threadIdx.x;      // 0 .. 4H*H-1
    if (idx >= G4H * H_) return;
    int gate = idx >> 20;                                  // H*H = 2^20
    int r    = idx & ((H_ * H_) - 1);
    const float* s = (gate == 0) ? Wg : (gate == 1) ? Wi : (gate == 2) ? Wf : Wo;
    Wb[idx] = (__bf16)s[r];
}

__global__ void pack_vec(const float* __restrict__ gx, const float* __restrict__ ix,
                         const float* __restrict__ fx, const float* __restrict__ ox,
                         const float* __restrict__ bg, const float* __restrict__ bi,
                         const float* __restrict__ bf, const float* __restrict__ bo,
                         float* __restrict__ Wx, float* __restrict__ bias) {
    int idx = blockIdx.x * blockDim.x + threadIdx.x;      // 0 .. 4H-1
    if (idx >= G4H) return;
    int gate = idx / H_;
    int r    = idx % H_;
    const float* sx = (gate == 0) ? gx : (gate == 1) ? ix : (gate == 2) ? fx : ox;
    const float* sb = (gate == 0) ? bg : (gate == 1) ? bi : (gate == 2) ? bf : bo;
    Wx[idx]   = sx[r];
    bias[idx] = sb[r];
}

__global__ void init_state(__bf16* __restrict__ h0, float* __restrict__ c) {
    int idx = blockIdx.x * blockDim.x + threadIdx.x;      // 0 .. B*H-1
    if (idx >= B_ * H_) return;
    h0[idx] = (__bf16)0.0f;
    c[idx]  = 0.0f;
}

// ---------------------------------------------------------------------------
// Stage h_in[bt*32 .. bt*32+31][k0 .. k0+KCHUNK) into one LDS A-tile buffer.
// 32 rows x 512 B = 16 KB per chunk = 1024 x 16B transfers; 128 threads issue
// 8 async b128 transfers each (per-lane LDS dest address, ASYNCcnt-tracked).
// ---------------------------------------------------------------------------
__device__ __forceinline__ void stage_chunk(const __bf16* __restrict__ h_in,
                                            __bf16 (*__restrict__ tile)[AROW],
                                            int bt, int k0, int tid) {
    const int r  = tid >> 2;                              // row 0..31
    const int c4 = tid & 3;                               // 0..3
    const __bf16* g = h_in + (size_t)(bt * 32 + r) * H_ + k0;
#pragma unroll
    for (int i = 0; i < 8; ++i) {
        const int col = (c4 + i * 4) * 8;                 // 8 bf16 = 16 B units
        ASYNC_COPY_B128(g + col, &tile[r][col]);
    }
}

// ---------------------------------------------------------------------------
// One LSTM time step.
//   z[b,n] = x[b,t]*Wx[n] + sum_k h[b,k]*Wb[n,k] + bias[n]   for n in [0,4H)
// Block = 128 threads (4 waves). Wave g owns gate g.
// Grid  = (H/16) j-tiles * (B/32) b-tiles = 64*8 = 512 blocks.
//
// The 32xK A-tile (h rows) is SHARED by all 4 gate-waves: it is staged into
// LDS once per block via async DMA, double-buffered over K-chunks of 256 so
// the DMA of chunk kc+1 overlaps WMMA on chunk kc. B (per-gate W_h rows)
// streams from global/L2 with prefetch.
//
// Fragment layouts used here (k contiguous in both A and B), wave32:
//   A 16x32 bf16 : lane L<16 -> M=L,   K=0..15
//                  lane L>=16-> M=L-16,K=16..31   (16B x2 per lane)
//   B 32x16 bf16 : lane L<16 -> N=L,   K=0..15
//                  lane L>=16-> N=L-16,K=16..31              (32B per lane)
//   C/D 16x16 f32: VGPR e, lane L<16 -> M=e,   N=L
//                          lane L>=16-> M=e+8, N=L-16
// ---------------------------------------------------------------------------
__global__ __launch_bounds__(128)
void lstm_step(const __bf16* __restrict__ Wb, const float* __restrict__ Wx,
               const float* __restrict__ bias, const float* __restrict__ x, int t,
               const __bf16* __restrict__ h_in, __bf16* __restrict__ h_out,
               float* __restrict__ c, float* __restrict__ h_f32) {
    __shared__ __bf16 Atile[2][32][AROW];                 // ~33 KB, double buf
    __shared__ float  zbuf[4][32][16];                    // 8 KB: z per gate

    const int tid   = threadIdx.x;
    const int wave  = tid >> 5;                           // gate id 0..3
    const int lane  = tid & 31;
    const int laneh = lane & 15;
    const int hi    = lane >> 4;                          // 0 or 1

    const int jt = blockIdx.x & 63;                       // hidden-col tile
    const int bt = blockIdx.x >> 6;                       // batch tile
    const int n0 = jt * 16;

    const int aoff = hi * 16;                             // K sub-offset for A
    const int rowW = wave * H_ + n0 + laneh;              // W_h row = out col
    const int boff = hi * 16;                             // K sub-offset for B
    const __bf16* pB = Wb + (size_t)rowW * H_ + boff;

    // Prologue: stage chunk 0.
    stage_chunk(h_in, Atile[0], bt, 0, tid);
    ASYNC_WAIT();
    __syncthreads();

    v8f acc0 = {};
    v8f acc1 = {};

    for (int kc = 0; kc < NCHUNK; ++kc) {
        const int cur = kc & 1;
        if (kc + 1 < NCHUNK) {                            // overlap DMA of next
            stage_chunk(h_in, Atile[cur ^ 1], bt, (kc + 1) * KCHUNK, tid);
        }
        const __bf16* lA0 = &Atile[cur][laneh][aoff];     // rows b0..b0+15
        const __bf16* lA1 = &Atile[cur][laneh + 16][aoff];// rows b0+16..b0+31
        const __bf16* pBk = pB + kc * KCHUNK;
        __builtin_prefetch(pBk + KCHUNK, 0, 1);           // global_prefetch_b8

#pragma unroll
        for (int kk = 0; kk < KCHUNK; kk += 32) {
            v8bf a0l = *(const v8bf*)(lA0 + kk);          // ds_load_b128
            v8bf a0h = *(const v8bf*)(lA0 + kk + 8);
            v8bf a1l = *(const v8bf*)(lA1 + kk);
            v8bf a1h = *(const v8bf*)(lA1 + kk + 8);
            v16bf bfrag = *(const v16bf*)(pBk + kk);      // global b128 x2

            v16bf A0 = __builtin_shufflevector(a0l, a0h, 0, 1, 2, 3, 4, 5, 6, 7,
                                               8, 9, 10, 11, 12, 13, 14, 15);
            v16bf A1 = __builtin_shufflevector(a1l, a1h, 0, 1, 2, 3, 4, 5, 6, 7,
                                               8, 9, 10, 11, 12, 13, 14, 15);

            acc0 = __builtin_amdgcn_wmma_f32_16x16x32_bf16(
                false, A0, false, bfrag, (short)0, acc0, false, false);
            acc1 = __builtin_amdgcn_wmma_f32_16x16x32_bf16(
                false, A1, false, bfrag, (short)0, acc1, false, false);
        }

        ASYNC_WAIT();                                     // next chunk landed
        __syncthreads();                                  // publish + reuse-safe
    }

    // Epilogue part 1: add bias + x-projection, scatter z into LDS.
    {
        const float bb = bias[wave * H_ + n0 + laneh];
        const float wx = Wx[wave * H_ + n0 + laneh];
#pragma unroll
        for (int e = 0; e < 8; ++e) {
            int m  = e + hi * 8;                          // rows 0..15 (acc0)
            int m2 = m + 16;                              // rows 16..31 (acc1)
            float xb  = x[(bt * 32 + m)  * T_ + t];
            float xb2 = x[(bt * 32 + m2) * T_ + t];
            zbuf[wave][m][laneh]  = acc0[e] + bb + xb * wx;
            zbuf[wave][m2][laneh] = acc1[e] + bb + xb2 * wx;
        }
    }
    __syncthreads();

    // Epilogue part 2: LSTM cell update for this (b-tile, j-tile).
    for (int s = tid; s < 32 * 16; s += 128) {
        int m  = s >> 4;
        int nn = s & 15;
        float zg = zbuf[0][m][nn];
        float zi = zbuf[1][m][nn];
        float zf = zbuf[2][m][nn];
        float zo = zbuf[3][m][nn];
        float g  = tanhf(zg);
        float ii = sigmoidf_(zi);
        float ff = sigmoidf_(zf);
        float oo = sigmoidf_(zo);
        size_t idx = (size_t)(bt * 32 + m) * H_ + n0 + nn;
        float cn = g * ii + c[idx] * ff;
        c[idx] = cn;
        float hn = tanhf(cn) * oo;
        h_out[idx] = (__bf16)hn;
        h_f32[idx] = hn;
    }
}

// ---------------------------------------------------------------------------
// Final projection: out[b,cc] = sum_k h[b,k] * W_ph[cc,k] + b_p[cc]
// One block per batch row; wave cc (10 waves) reduces one output class.
// ---------------------------------------------------------------------------
__global__ __launch_bounds__(320)
void proj(const float* __restrict__ h, const float* __restrict__ Wp,
          const float* __restrict__ bp, float* __restrict__ out) {
    int b    = blockIdx.x;
    int cc   = threadIdx.x >> 5;                          // 0..9
    int lane = threadIdx.x & 31;
    float acc = 0.0f;
    for (int k = lane; k < H_; k += 32) {
        acc += h[b * H_ + k] * Wp[cc * H_ + k];
    }
#pragma unroll
    for (int off = 16; off > 0; off >>= 1) {
        acc += __shfl_down(acc, off, 32);
    }
    if (lane == 0) {
        out[b * C_ + cc] = acc + bp[cc];
    }
}

// ---------------------------------------------------------------------------
extern "C" void kernel_launch(void* const* d_in, const int* in_sizes, int n_in,
                              void* d_out, int out_size, void* d_ws, size_t ws_size,
                              hipStream_t stream) {
    const float* x    = (const float*)d_in[0];
    const float* Wgx  = (const float*)d_in[1];
    const float* Wgh  = (const float*)d_in[2];
    const float* bg   = (const float*)d_in[3];
    const float* Wix  = (const float*)d_in[4];
    const float* Wih  = (const float*)d_in[5];
    const float* bi   = (const float*)d_in[6];
    const float* Wfx  = (const float*)d_in[7];
    const float* Wfh  = (const float*)d_in[8];
    const float* bf   = (const float*)d_in[9];
    const float* Wox  = (const float*)d_in[10];
    const float* Woh  = (const float*)d_in[11];
    const float* bo   = (const float*)d_in[12];
    const float* Wph  = (const float*)d_in[13];
    const float* bp   = (const float*)d_in[14];
    float* out = (float*)d_out;

    // Workspace carve-up (all chunks naturally 256B-aligned).
    char* ws = (char*)d_ws;
    size_t off = 0;
    __bf16* Wb   = (__bf16*)(ws + off); off += (size_t)G4H * H_ * sizeof(__bf16); // 8 MB
    float*  Wx   = (float*)(ws + off);  off += (size_t)G4H * sizeof(float);
    float*  bias = (float*)(ws + off);  off += (size_t)G4H * sizeof(float);
    __bf16* h0   = (__bf16*)(ws + off); off += (size_t)B_ * H_ * sizeof(__bf16);
    __bf16* h1   = (__bf16*)(ws + off); off += (size_t)B_ * H_ * sizeof(__bf16);
    float*  c    = (float*)(ws + off);  off += (size_t)B_ * H_ * sizeof(float);
    float*  hf   = (float*)(ws + off);  off += (size_t)B_ * H_ * sizeof(float);
    (void)ws_size; (void)in_sizes; (void)n_in; (void)out_size;

    // One-time packing + state init (inside graph => deterministic per call).
    pack_wh<<<(G4H * H_ + 255) / 256, 256, 0, stream>>>(Wgh, Wih, Wfh, Woh, Wb);
    pack_vec<<<(G4H + 255) / 256, 256, 0, stream>>>(Wgx, Wix, Wfx, Wox,
                                                    bg, bi, bf, bo, Wx, bias);
    init_state<<<(B_ * H_ + 255) / 256, 256, 0, stream>>>(h0, c);

    // Serial scan: stream ordering provides the step-to-step dependency.
    for (int t = 0; t < T_; ++t) {
        const __bf16* hin = (t & 1) ? h1 : h0;
        __bf16* hout      = (t & 1) ? h0 : h1;
        lstm_step<<<dim3(512), dim3(128), 0, stream>>>(Wb, Wx, bias, x, t,
                                                       hin, hout, c, hf);
    }

    proj<<<dim3(B_), dim3(320), 0, stream>>>(hf, Wph, bp, out);
}
